// TransformerHead_13288628813909
// MI455X (gfx1250) — compile-verified, matrix operand order checked by reading, not run on hardware
//
#include <hip/hip_runtime.h>

// ---------------------------------------------------------------------------
// Problem constants (reference: B,S,E,H,D = 4,2048,1024,8,256)
// ---------------------------------------------------------------------------
#define B_ 4
#define S_ 2048
#define E_ 1024
#define H_ 8
#define D_ 256
#define BS_ (B_ * S_)      // 8192
#define HD_ (H_ * D_)      // 2048

typedef __bf16 bf16_t;
typedef __attribute__((ext_vector_type(16))) __bf16 bf16x16;
typedef __attribute__((ext_vector_type(8)))  float  floatx8;
typedef __attribute__((ext_vector_type(4)))  unsigned int uint4v;

// float -> bf16, round-to-nearest-even (bit trick; avoids any libcall risk)
__device__ __forceinline__ bf16_t f2bf(float x) {
  unsigned u = __float_as_uint(x);
  u += 0x7FFFu + ((u >> 16) & 1u);
  unsigned short s = (unsigned short)(u >> 16);
  return __builtin_bit_cast(bf16_t, s);
}

// ---------------------------------------------------------------------------
// CDNA5 async copy: global -> LDS, 16B per lane, tracked by ASYNCcnt.
//   LDS address from the lds_off VGPR; global address from a 64-bit VGPR pair
// LDS offset = low 32 bits of the generic pointer (flat LDS aperture keeps
// the byte offset in addr[31:0]).
// ---------------------------------------------------------------------------
__device__ __forceinline__ void async_copy_b128(unsigned lds_off,
                                                const bf16_t* gaddr) {
  asm volatile("global_load_async_to_lds_b128 %0, %1, off"
               :: "v"(lds_off), "v"(gaddr) : "memory");
}

// ---------------------------------------------------------------------------
// WMMA fragment load (from LDS or global).
// 16-bit 16x32 A/B fragment (wave32): lane (l16, half):
//   elements 0..7  <-> K = kbase + 8*half + e
//   elements 8..15 <-> K = kbase + 16 + 8*half + (e-8)
// Caller passes p = row_ptr + kbase + 8*half; we load p[0..7] and p[16..23].
// ---------------------------------------------------------------------------
__device__ __forceinline__ bf16x16 load_frag(const bf16_t* __restrict__ p) {
  union { bf16x16 v; uint4v u[2]; } f;
  f.u[0] = *(const uint4v*)(p);
  f.u[1] = *(const uint4v*)(p + 16);
  return f.v;
}

// ---------------------------------------------------------------------------
// LDS-staged, double-buffered block GEMM inner loop.
//   Block = 256 threads = 8 wave32s; block tile 128(M) x 64(N); k-stage 64.
//   A row-major [M][K] (lda); Bt row-major [N][K] (ldb)  (B pre-transposed).
//   Wave (wm = wid&3, wn = wid>>2) owns a 32x32 tile = 2x2 WMMA subtiles.
//   Stage fill: 6 global_load_async_to_lds_b128 per thread (4 A + 2 B).
//   LDS rows padded to 72 bf16 (144B) for bank-conflict-free b128 reads.
// ---------------------------------------------------------------------------
#define KS_  64   // K elements per stage
#define AST_ 72   // LDS row stride in elements (64 + 8 pad)

__device__ __forceinline__ void wmma_block_gemm(
    const bf16_t* __restrict__ Ablock, size_t lda,
    const bf16_t* __restrict__ Btblock, size_t ldb, int Kd,
    floatx8& c00, floatx8& c01, floatx8& c10, floatx8& c11) {
  __shared__ __attribute__((aligned(16))) bf16_t Asmem[2][128 * AST_];
  __shared__ __attribute__((aligned(16))) bf16_t Bsmem[2][64 * AST_];

  const int tid  = threadIdx.x;
  const int lane = tid & 31, l16 = lane & 15, half = lane >> 4;
  const int wid  = tid >> 5, wm = wid & 3, wn = wid >> 2;

  const unsigned a_off = (unsigned)(size_t)(&Asmem[0][0]);
  const unsigned b_off = (unsigned)(size_t)(&Bsmem[0][0]);

  // Stage k-slice [k0, k0+64) of the A and B bands into LDS buffer `buf`.
  auto stage = [&](int k0, int buf) {
    const unsigned ab = a_off + (unsigned)buf * (128 * AST_ * 2);
    #pragma unroll
    for (int j = 0; j < 4; ++j) {                    // 128 rows x 8 chunks
      const int c = tid + 256 * j;
      const int row = c >> 3, col = (c & 7) * 8;
      async_copy_b128(ab + (unsigned)(row * AST_ + col) * 2,
                      Ablock + (size_t)row * lda + k0 + col);
    }
    const unsigned bb = b_off + (unsigned)buf * (64 * AST_ * 2);
    #pragma unroll
    for (int j = 0; j < 2; ++j) {                    // 64 rows x 8 chunks
      const int c = tid + 256 * j;
      const int row = c >> 3, col = (c & 7) * 8;
      async_copy_b128(bb + (unsigned)(row * AST_ + col) * 2,
                      Btblock + (size_t)row * ldb + k0 + col);
    }
  };

  const int nst = Kd / KS_;
  stage(0, 0);
  for (int st = 0; st < nst; ++st) {
    const int cur = st & 1;
    if (st + 1 < nst) {
      stage((st + 1) * KS_, cur ^ 1);
      // async loads retire in order: <=6 outstanding => current stage landed
      asm volatile("s_wait_asynccnt 0x6" ::: "memory");
    } else {
      asm volatile("s_wait_asynccnt 0x0" ::: "memory");
    }
    __syncthreads();

    #pragma unroll
    for (int kk = 0; kk < KS_; kk += 32) {
      const bf16_t* A0 = &Asmem[cur][(wm * 32 + l16) * AST_ + kk + 8 * half];
      const bf16_t* B0 = &Bsmem[cur][(wn * 32 + l16) * AST_ + kk + 8 * half];
      bf16x16 fa0 = load_frag(A0);
      bf16x16 fa1 = load_frag(A0 + 16 * AST_);
      bf16x16 fb0 = load_frag(B0);
      bf16x16 fb1 = load_frag(B0 + 16 * AST_);
      c00 = __builtin_amdgcn_wmma_f32_16x16x32_bf16(false, fa0, false, fb0, (short)0, c00, false, false);
      c01 = __builtin_amdgcn_wmma_f32_16x16x32_bf16(false, fa0, false, fb1, (short)0, c01, false, false);
      c10 = __builtin_amdgcn_wmma_f32_16x16x32_bf16(false, fa1, false, fb0, (short)0, c10, false, false);
      c11 = __builtin_amdgcn_wmma_f32_16x16x32_bf16(false, fa1, false, fb1, (short)0, c11, false, false);
    }
    __syncthreads();   // all waves done reading before buffer is re-filled
  }
}

// ---------------------------------------------------------------------------
// Conversion kernels
// ---------------------------------------------------------------------------
__global__ __launch_bounds__(256) void k_cvt_bf16(const float* __restrict__ src,
                                                  bf16_t* __restrict__ dst, int n) {
  int i = (blockIdx.x * 256 + threadIdx.x) * 4;
  if (i + 3 < n) {
    float4 v = *(const float4*)(src + i);
    dst[i + 0] = f2bf(v.x);
    dst[i + 1] = f2bf(v.y);
    dst[i + 2] = f2bf(v.z);
    dst[i + 3] = f2bf(v.w);
  }
}

// src: [z][rows][cols] f32  ->  dst: [z][cols][rows] bf16   (32x32 LDS tiles)
__global__ __launch_bounds__(256) void k_cvt_transpose(const float* __restrict__ src,
                                                       bf16_t* __restrict__ dst,
                                                       int rows, int cols) {
  __shared__ float tile[32][33];
  const int z = blockIdx.z;
  const float* s = src + (size_t)z * rows * cols;
  bf16_t*      d = dst + (size_t)z * rows * cols;
  const int r0 = blockIdx.y * 32;
  const int c0 = blockIdx.x * 32;
  for (int i = threadIdx.y; i < 32; i += 8)
    tile[i][threadIdx.x] = s[(size_t)(r0 + i) * cols + (c0 + threadIdx.x)];
  __syncthreads();
  for (int i = threadIdx.y; i < 32; i += 8)
    d[(size_t)(c0 + i) * rows + (r0 + threadIdx.x)] = f2bf(tile[threadIdx.x][i]);
}

// ---------------------------------------------------------------------------
// QKV projection GEMM + bias + ReLU.
//   A16:[BS][E] bf16, Wt:[H][D][E] bf16 (pre-transposed), out per head.
//   blockIdx.z = p*H + h  (p: 0=Q, 1=K, 2=V)
//   Q16/K16: [b][h][s][d] bf16;  V stored transposed Vt16: [b][h][d][s] bf16.
// ---------------------------------------------------------------------------
__global__ __launch_bounds__(256) void k_gemm_qkv(
    const bf16_t* __restrict__ A16,
    const bf16_t* __restrict__ WQt, const bf16_t* __restrict__ WKt,
    const bf16_t* __restrict__ WVt,
    const float* __restrict__ bQ, const float* __restrict__ bK,
    const float* __restrict__ bV,
    bf16_t* __restrict__ Q16, bf16_t* __restrict__ K16,
    bf16_t* __restrict__ Vt16) {
  const int z = blockIdx.z;
  const int p = z >> 3;
  const int h = z & 7;
  const bf16_t* Wt  = (p == 0 ? WQt : (p == 1 ? WKt : WVt)) + (size_t)h * D_ * E_;
  const float* bias = (p == 0 ? bQ  : (p == 1 ? bK  : bV )) + h * D_;

  const int wid = threadIdx.x >> 5;
  const int m_base = blockIdx.x * 128 + (wid & 3) * 32;
  const int n_base = blockIdx.y * 64 + (wid >> 2) * 32;

  floatx8 c00 = {0,0,0,0,0,0,0,0}, c01 = c00, c10 = c00, c11 = c00;
  wmma_block_gemm(A16 + (size_t)blockIdx.x * 128 * E_, E_,
                  Wt  + (size_t)blockIdx.y * 64 * E_, E_, E_,
                  c00, c01, c10, c11);

  const int lane = threadIdx.x & 31, l16 = lane & 15, half = lane >> 4;
  const int b = m_base / S_;                 // tile never straddles batches
  for (int mi = 0; mi < 2; ++mi)
    for (int ni = 0; ni < 2; ++ni) {
      floatx8 acc = (mi == 0) ? (ni == 0 ? c00 : c01) : (ni == 0 ? c10 : c11);
      const int n = n_base + ni * 16 + l16;  // d index
      const float bv = bias[n];
      for (int r = 0; r < 8; ++r) {
        const int m = m_base + mi * 16 + 8 * half + r;
        const int s = m - b * S_;
        float v = acc[r] + bv;
        v = fmaxf(v, 0.0f);
        const size_t bh = (size_t)b * H_ + h;
        if (p == 0)      Q16[(bh * S_ + s) * D_ + n] = f2bf(v);
        else if (p == 1) K16[(bh * S_ + s) * D_ + n] = f2bf(v);
        else             Vt16[(bh * D_ + n) * S_ + s] = f2bf(v);
      }
    }
}

// ---------------------------------------------------------------------------
// scores[s][t] = (1/32) * sum_d Q[s][d]*K[t][d]    (NT GEMM, per b,h)
// ---------------------------------------------------------------------------
__global__ __launch_bounds__(256) void k_gemm_scores(
    const bf16_t* __restrict__ Q16, const bf16_t* __restrict__ K16,
    float* __restrict__ SC) {
  const int bh = blockIdx.z;
  const bf16_t* Qh = Q16 + (size_t)bh * S_ * D_;
  const bf16_t* Kh = K16 + (size_t)bh * S_ * D_;
  float* SCh = SC + (size_t)bh * S_ * S_;

  const int wid = threadIdx.x >> 5;
  const int m_base = blockIdx.x * 128 + (wid & 3) * 32;
  const int n_base = blockIdx.y * 64 + (wid >> 2) * 32;

  floatx8 c00 = {0,0,0,0,0,0,0,0}, c01 = c00, c10 = c00, c11 = c00;
  wmma_block_gemm(Qh + (size_t)blockIdx.x * 128 * D_, D_,
                  Kh + (size_t)blockIdx.y * 64 * D_, D_, D_,
                  c00, c01, c10, c11);

  const int lane = threadIdx.x & 31, l16 = lane & 15, half = lane >> 4;
  for (int mi = 0; mi < 2; ++mi)
    for (int ni = 0; ni < 2; ++ni) {
      floatx8 acc = (mi == 0) ? (ni == 0 ? c00 : c01) : (ni == 0 ? c10 : c11);
      const int t = n_base + ni * 16 + l16;
      for (int r = 0; r < 8; ++r) {
        const int s = m_base + mi * 16 + 8 * half + r;
        SCh[(size_t)s * S_ + t] = acc[r] * 0.03125f;   // 1/sqrt(E)
      }
    }
}

// ---------------------------------------------------------------------------
// Softmax over the query axis s (column softmax), in place on SC.
// block(32,8): 32 columns per block, 8-way row strip-mining, LDS reduce.
// ---------------------------------------------------------------------------
__global__ __launch_bounds__(256) void k_softmax_s(float* __restrict__ SC) {
  __shared__ float red[8][32];
  const int bh = blockIdx.z;
  const int t  = blockIdx.x * 32 + threadIdx.x;
  float* M = SC + (size_t)bh * S_ * S_;

  float mx = -1e30f;
  for (int s = threadIdx.y; s < S_; s += 8)
    mx = fmaxf(mx, M[(size_t)s * S_ + t]);
  red[threadIdx.y][threadIdx.x] = mx;
  __syncthreads();
  if (threadIdx.y == 0) {
    float v = red[0][threadIdx.x];
    for (int i = 1; i < 8; ++i) v = fmaxf(v, red[i][threadIdx.x]);
    red[0][threadIdx.x] = v;
  }
  __syncthreads();
  mx = red[0][threadIdx.x];
  __syncthreads();

  float sum = 0.0f;
  for (int s = threadIdx.y; s < S_; s += 8) {
    float e = __expf(M[(size_t)s * S_ + t] - mx);
    M[(size_t)s * S_ + t] = e;
    sum += e;
  }
  red[threadIdx.y][threadIdx.x] = sum;
  __syncthreads();
  if (threadIdx.y == 0) {
    float v = 0.0f;
    for (int i = 0; i < 8; ++i) v += red[i][threadIdx.x];
    red[0][threadIdx.x] = v;
  }
  __syncthreads();
  const float inv = 1.0f / red[0][threadIdx.x];
  for (int s = threadIdx.y; s < S_; s += 8)
    M[(size_t)s * S_ + t] *= inv;
}

// ---------------------------------------------------------------------------
// Softmax over the key axis t (row softmax); writes bf16 probs to P16.
// One 256-thread block per row; each thread owns 8 elements.
// ---------------------------------------------------------------------------
__global__ __launch_bounds__(256) void k_softmax_t(const float* __restrict__ SC,
                                                   bf16_t* __restrict__ P16) {
  __shared__ float red[256];
  const int bh = blockIdx.y;
  const int s  = blockIdx.x;
  const float* row = SC + ((size_t)bh * S_ + s) * S_;
  bf16_t*     prow = P16 + ((size_t)bh * S_ + s) * S_;

  float x[8];
  float mx = -1e30f;
  for (int i = 0; i < 8; ++i) {
    x[i] = row[threadIdx.x + 256 * i];
    mx = fmaxf(mx, x[i]);
  }
  red[threadIdx.x] = mx;
  __syncthreads();
  for (int o = 128; o > 0; o >>= 1) {
    if (threadIdx.x < o) red[threadIdx.x] = fmaxf(red[threadIdx.x], red[threadIdx.x + o]);
    __syncthreads();
  }
  mx = red[0];
  __syncthreads();

  float sum = 0.0f;
  for (int i = 0; i < 8; ++i) {
    x[i] = __expf(x[i] - mx);
    sum += x[i];
  }
  red[threadIdx.x] = sum;
  __syncthreads();
  for (int o = 128; o > 0; o >>= 1) {
    if (threadIdx.x < o) red[threadIdx.x] += red[threadIdx.x + o];
    __syncthreads();
  }
  const float inv = 1.0f / red[0];
  for (int i = 0; i < 8; ++i)
    prow[threadIdx.x + 256 * i] = f2bf(x[i] * inv);
}

// ---------------------------------------------------------------------------
// Z = A(probs) @ V, per (b,h).  A=P16 [s][t] row-major; B=Vt16 [d][t] (so
// B-fragment loads are contiguous).  Output into concat layout Z16 [b][s][h*D+d].
// ---------------------------------------------------------------------------
__global__ __launch_bounds__(256) void k_gemm_av(
    const bf16_t* __restrict__ P16, const bf16_t* __restrict__ Vt16,
    bf16_t* __restrict__ Z16) {
  const int bh = blockIdx.z;
  const int b = bh >> 3, h = bh & 7;
  const bf16_t* Ph = P16  + (size_t)bh * S_ * S_;
  const bf16_t* Vh = Vt16 + (size_t)bh * D_ * S_;

  const int wid = threadIdx.x >> 5;
  const int m_base = blockIdx.x * 128 + (wid & 3) * 32;
  const int n_base = blockIdx.y * 64 + (wid >> 2) * 32;

  floatx8 c00 = {0,0,0,0,0,0,0,0}, c01 = c00, c10 = c00, c11 = c00;
  wmma_block_gemm(Ph + (size_t)blockIdx.x * 128 * S_, S_,
                  Vh + (size_t)blockIdx.y * 64 * S_, S_, S_,
                  c00, c01, c10, c11);

  const int lane = threadIdx.x & 31, l16 = lane & 15, half = lane >> 4;
  for (int mi = 0; mi < 2; ++mi)
    for (int ni = 0; ni < 2; ++ni) {
      floatx8 acc = (mi == 0) ? (ni == 0 ? c00 : c01) : (ni == 0 ? c10 : c11);
      const int d = n_base + ni * 16 + l16;
      for (int r = 0; r < 8; ++r) {
        const int s = m_base + mi * 16 + 8 * half + r;
        Z16[((size_t)b * S_ + s) * HD_ + h * D_ + d] = f2bf(acc[r]);
      }
    }
}

// ---------------------------------------------------------------------------
// out = Z16[BS][HD] @ WZ[HD][E] + bZ   (B pre-transposed: WZt [E][HD]) -> f32
// ---------------------------------------------------------------------------
__global__ __launch_bounds__(256) void k_gemm_out(
    const bf16_t* __restrict__ Z16, const bf16_t* __restrict__ WZt,
    const float* __restrict__ bZ, float* __restrict__ out) {
  const int wid = threadIdx.x >> 5;
  const int m_base = blockIdx.x * 128 + (wid & 3) * 32;
  const int n_base = blockIdx.y * 64 + (wid >> 2) * 32;

  floatx8 c00 = {0,0,0,0,0,0,0,0}, c01 = c00, c10 = c00, c11 = c00;
  wmma_block_gemm(Z16 + (size_t)blockIdx.x * 128 * HD_, HD_,
                  WZt + (size_t)blockIdx.y * 64 * HD_, HD_, HD_,
                  c00, c01, c10, c11);

  const int lane = threadIdx.x & 31, l16 = lane & 15, half = lane >> 4;
  for (int mi = 0; mi < 2; ++mi)
    for (int ni = 0; ni < 2; ++ni) {
      floatx8 acc = (mi == 0) ? (ni == 0 ? c00 : c01) : (ni == 0 ? c10 : c11);
      const int n = n_base + ni * 16 + l16;
      const float bv = bZ[n];
      for (int r = 0; r < 8; ++r) {
        const int m = m_base + mi * 16 + 8 * half + r;
        out[(size_t)m * E_ + n] = acc[r] + bv;
      }
    }
}

// ---------------------------------------------------------------------------
// Host launcher
// ---------------------------------------------------------------------------
extern "C" void kernel_launch(void* const* d_in, const int* in_sizes, int n_in,
                              void* d_out, int out_size, void* d_ws, size_t ws_size,
                              hipStream_t stream) {
  const float* inA = (const float*)d_in[0];
  const float* WQ  = (const float*)d_in[1];
  const float* bQ  = (const float*)d_in[2];
  const float* WK  = (const float*)d_in[3];
  const float* bK  = (const float*)d_in[4];
  const float* WV  = (const float*)d_in[5];
  const float* bV  = (const float*)d_in[6];
  const float* WZ  = (const float*)d_in[7];
  const float* bZ  = (const float*)d_in[8];
  float* out = (float*)d_out;

  // Workspace carve-up (all sizes 256B aligned by construction)
  char* ws = (char*)d_ws;
  size_t off = 0;
  auto carve = [&](size_t bytes) { void* p = ws + off; off += (bytes + 255) & ~(size_t)255; return p; };
  bf16_t* A16  = (bf16_t*)carve((size_t)BS_ * E_ * 2);       // 16.8 MB
  bf16_t* WQt  = (bf16_t*)carve((size_t)H_ * D_ * E_ * 2);   // 4.2 MB
  bf16_t* WKt  = (bf16_t*)carve((size_t)H_ * D_ * E_ * 2);
  bf16_t* WVt  = (bf16_t*)carve((size_t)H_ * D_ * E_ * 2);
  bf16_t* WZt  = (bf16_t*)carve((size_t)E_ * HD_ * 2);       // 4.2 MB
  bf16_t* Q16  = (bf16_t*)carve((size_t)B_ * H_ * S_ * D_ * 2); // 33.6 MB
  bf16_t* K16  = (bf16_t*)carve((size_t)B_ * H_ * S_ * D_ * 2);
  bf16_t* Vt16 = (bf16_t*)carve((size_t)B_ * H_ * S_ * D_ * 2);
  bf16_t* Z16  = (bf16_t*)carve((size_t)BS_ * HD_ * 2);      // 33.6 MB
  float*  SC   = (float*) carve((size_t)B_ * H_ * S_ * S_ * 4); // 537 MB
  bf16_t* P16  = (bf16_t*)carve((size_t)B_ * H_ * S_ * S_ * 2); // 268 MB

  // 1) bf16 conversions + weight transposes
  {
    int n = BS_ * E_;
    k_cvt_bf16<<<dim3(n / (256 * 4)), dim3(256), 0, stream>>>(inA, A16, n);
  }
  k_cvt_transpose<<<dim3(D_ / 32, E_ / 32, H_), dim3(32, 8), 0, stream>>>(WQ, WQt, E_, D_);
  k_cvt_transpose<<<dim3(D_ / 32, E_ / 32, H_), dim3(32, 8), 0, stream>>>(WK, WKt, E_, D_);
  k_cvt_transpose<<<dim3(D_ / 32, E_ / 32, H_), dim3(32, 8), 0, stream>>>(WV, WVt, E_, D_);
  k_cvt_transpose<<<dim3(E_ / 32, HD_ / 32, 1), dim3(32, 8), 0, stream>>>(WZ, WZt, HD_, E_);

  // 2) Q/K/V projections (+bias, ReLU); V stored transposed
  k_gemm_qkv<<<dim3(BS_ / 128, D_ / 64, 3 * H_), dim3(256), 0, stream>>>(
      A16, WQt, WKt, WVt, bQ, bK, bV, Q16, K16, Vt16);

  // 3) scores = Q K^T / 32
  k_gemm_scores<<<dim3(S_ / 128, S_ / 64, B_ * H_), dim3(256), 0, stream>>>(Q16, K16, SC);

  // 4) softmax over s (columns), in place
  k_softmax_s<<<dim3(S_ / 32, 1, B_ * H_), dim3(32, 8), 0, stream>>>(SC);

  // 5) softmax over t (rows) -> bf16 probs
  k_softmax_t<<<dim3(S_, B_ * H_), dim3(256), 0, stream>>>(SC, P16);

  // 6) Z = P @ V  -> concat-head layout
  k_gemm_av<<<dim3(S_ / 128, D_ / 64, B_ * H_), dim3(256), 0, stream>>>(P16, Vt16, Z16);

  // 7) out = Z @ WZ + bZ  (f32)
  k_gemm_out<<<dim3(BS_ / 128, E_ / 64, 1), dim3(256), 0, stream>>>(Z16, WZt, bZ, out);
}
